// MHSA_RPB_20959440404505
// MI455X (gfx1250) — matrix operand order repaired and checked by reading, not compiled, not run on hardware
//
#include <hip/hip_runtime.h>

typedef __bf16 bf16;
typedef __attribute__((ext_vector_type(16))) __bf16 v16bf;
typedef __attribute__((ext_vector_type(8)))  float  v8f;
typedef __attribute__((ext_vector_type(4)))  unsigned int v4u;

union Frag {
    v16bf v;
    v4u   u[2];
    bf16  h[16];
};

__device__ inline v8f wmma_bf16(const Frag& a, const Frag& b, v8f c) {
    return __builtin_amdgcn_wmma_f32_16x16x32_bf16(false, a.v, false, b.v,
                                                   (short)0, c, false, false);
}

// Async global -> LDS copy of 16 bytes (GLOBAL_LOAD_ASYNC_TO_LDS_B128).
// LDS byte address = low 32 bits of the generic shared pointer (ISA 10.2).
__device__ inline void async_cp_b128(const bf16* gsrc, bf16* ldst) {
    unsigned lds_addr = (unsigned)(size_t)ldst;
    unsigned long long gaddr = (unsigned long long)(size_t)gsrc;
    asm volatile("global_load_async_to_lds_b128 %0, %1, off"
                 :: "v"(lds_addr), "v"(gaddr) : "memory");
}

__device__ inline void wait_async0() {
    asm volatile("s_wait_asynccnt 0" ::: "memory");
}

// ---------------------------------------------------------------------------
// f32 -> bf16 conversion (vector-of-4 grid stride)
// ---------------------------------------------------------------------------
__global__ __launch_bounds__(256) void cvt_f32_bf16(const float* __restrict__ in,
                                                    bf16* __restrict__ out, int n4) {
    union { bf16 h[4]; uint2 u; } pk;
    int stride = gridDim.x * blockDim.x;
    for (int i = blockIdx.x * blockDim.x + threadIdx.x; i < n4; i += stride) {
        float4 f = ((const float4*)in)[i];
        pk.h[0] = (bf16)f.x; pk.h[1] = (bf16)f.y;
        pk.h[2] = (bf16)f.z; pk.h[3] = (bf16)f.w;
        ((uint2*)out)[i] = pk.u;
    }
}

// ---------------------------------------------------------------------------
// biasH[h][q][k] = bias_table[rel_index[q*L+k]*H + h]
// ---------------------------------------------------------------------------
__global__ __launch_bounds__(256) void build_bias(const float* __restrict__ table,
                                                  const int* __restrict__ rel,
                                                  float* __restrict__ biasH) {
    int i = blockIdx.x * 256 + threadIdx.x;   // over L*L = 65536
    int idx = rel[i];
#pragma unroll
    for (int h = 0; h < 16; ++h)
        biasH[(size_t)h * 65536 + i] = table[idx * 16 + h];
}

// ---------------------------------------------------------------------------
// Tiled WMMA GEMM:  C[M,N] = A[M,K] * W[N,K]^T + bias[N]
// block tile 64(M) x 256(N); 8 waves (2x4), each 32x64 = 2x4 WMMA C-tiles.
// Double-buffered async global->LDS staging (ASYNCcnt).
// ---------------------------------------------------------------------------
template <bool OUT_BF16>
__global__ __launch_bounds__(256) void gemm_wmma(const bf16* __restrict__ A,
                                                 const bf16* __restrict__ W,
                                                 const float* __restrict__ bias,
                                                 void* __restrict__ out,
                                                 int M, int N, int K) {
    __shared__ bf16 As[2][64 * 32];
    __shared__ bf16 Ws[2][256 * 32];

    const int nBlk = N >> 8;
    const int bm = blockIdx.x / nBlk;
    const int bn = blockIdx.x % nBlk;
    const int tid = threadIdx.x;
    const int wave = tid >> 5;
    const int lane = tid & 31;
    const int wm = wave >> 2;       // 0..1
    const int wn = wave & 3;        // 0..3
    const int m16 = lane & 15;

    v8f acc[2][4] = {};

    const int arow = tid >> 2, acol = (tid & 3) * 8;   // A coop-load map
    const bf16* agp = A + (size_t)(bm * 64 + arow) * K + acol;
    const bf16* wgp = W + (size_t)(bn * 256 + tid) * K;

    // issue async copies of K-step k0 into LDS stage s
    auto stage = [&](int k0, int s) {
        async_cp_b128(agp + k0, &As[s][arow * 32 + acol]);
        const bf16* src = wgp + k0;
        bf16* dst = &Ws[s][tid * 32];
        async_cp_b128(src,      dst);
        async_cp_b128(src + 8,  dst + 8);
        async_cp_b128(src + 16, dst + 16);
        async_cp_b128(src + 24, dst + 24);
    };

    stage(0, 0);

    int s = 0;
    for (int k0 = 0; k0 < K; k0 += 32, s ^= 1) {
        wait_async0();
        __syncthreads();
        if (k0 + 32 < K) stage(k0 + 32, s ^ 1);

        Frag a[2], b[4];
        const int aofs = (lane < 16) ? 0 : 16;    // A-layout K offset
        const int kofs = (lane < 16) ? 0 : 16;    // B-layout K offset
#pragma unroll
        for (int mt = 0; mt < 2; ++mt) {
            const bf16* p = &As[s][(wm * 32 + mt * 16 + m16) * 32 + aofs];
            a[mt].u[0] = *(const v4u*)(p);
            a[mt].u[1] = *(const v4u*)(p + 8);
        }
#pragma unroll
        for (int nt = 0; nt < 4; ++nt) {
            const bf16* p = &Ws[s][(wn * 64 + nt * 16 + m16) * 32 + kofs];
            b[nt].u[0] = *(const v4u*)(p);
            b[nt].u[1] = *(const v4u*)(p + 8);
        }
#pragma unroll
        for (int mt = 0; mt < 2; ++mt)
#pragma unroll
            for (int nt = 0; nt < 4; ++nt)
                acc[mt][nt] = wmma_bf16(a[mt], b[nt], acc[mt][nt]);
    }

    // epilogue
#pragma unroll
    for (int mt = 0; mt < 2; ++mt) {
#pragma unroll
        for (int nt = 0; nt < 4; ++nt) {
            const int ncol = bn * 256 + wn * 64 + nt * 16 + m16;
            const float bval = bias ? bias[ncol] : 0.0f;
            const int mbase = bm * 64 + wm * 32 + mt * 16 + ((lane < 16) ? 0 : 8);
            if (OUT_BF16) {
                bf16* o = (bf16*)out;
#pragma unroll
                for (int r = 0; r < 8; ++r)
                    o[(size_t)(mbase + r) * N + ncol] = (bf16)(acc[mt][nt][r] + bval);
            } else {
                float* o = (float*)out;
#pragma unroll
                for (int r = 0; r < 8; ++r)
                    o[(size_t)(mbase + r) * N + ncol] = acc[mt][nt][r] + bval;
            }
        }
    }
}

// ---------------------------------------------------------------------------
// Fused window attention, register-resident two-pass softmax.
// One workgroup per (b,h). qkv: [B*L, 1536] bf16 (q|k|v each 512, head h at h*32)
// biasH: [H, 256, 256] f32.   attn_out: [B*L, 512] bf16.
// Each wave owns two 16-row query tiles; full 16x256 score block kept in VGPRs.
// ---------------------------------------------------------------------------
__global__ __launch_bounds__(256) void attn_wmma(const bf16* __restrict__ qkv,
                                                 const float* __restrict__ biasH,
                                                 bf16* __restrict__ attn_out) {
    __shared__ bf16 Klds[256 * 32];      // [key][d]           16 KB
    __shared__ bf16 VT[32 * 256];        // [d][key]            16 KB
    __shared__ bf16 Pscr[8][16 * 256];   // per-wave P block    64 KB

    const int bh = blockIdx.x;
    const int b = bh >> 4;
    const int h = bh & 15;
    const int tid = threadIdx.x;
    const int wave = tid >> 5;
    const int lane = tid & 31;
    const int m16 = lane & 15;
    const bf16* base = qkv + (size_t)b * 256 * 1536;

    // stage K via async copies; V transposed through VGPRs
    {
        const bf16* ks = base + (size_t)tid * 1536 + 512 + h * 32;
        bf16* kd = Klds + tid * 32;
        async_cp_b128(ks,      kd);
        async_cp_b128(ks + 8,  kd + 8);
        async_cp_b128(ks + 16, kd + 16);
        async_cp_b128(ks + 24, kd + 24);
        const bf16* vs = base + (size_t)tid * 1536 + 1024 + h * 32;
#pragma unroll
        for (int d = 0; d < 32; ++d)
            VT[d * 256 + tid] = vs[d];
    }
    wait_async0();
    __syncthreads();

    const float scale = 0.17677669529663687f;   // 1/sqrt(32)

    for (int qt = wave; qt < 16; qt += 8) {
        const int qbase = qt * 16;
        Frag qf;
        {
            const bf16* qs = base + (size_t)(qbase + m16) * 1536 + h * 32 +
                             ((lane < 16) ? 0 : 16);
            qf.u[0] = *(const v4u*)(qs);
            qf.u[1] = *(const v4u*)(qs + 8);
        }

        // ---- phase 1: all 16 QK^T WMMAs, scale+bias fused ----
        v8f sc[16];
        const int dofs = (lane < 16) ? 0 : 16;
#pragma unroll
        for (int t = 0; t < 16; ++t) {
            Frag kf;
            const bf16* kp = Klds + (t * 16 + m16) * 32 + dofs;
            kf.u[0] = *(const v4u*)(kp);
            kf.u[1] = *(const v4u*)(kp + 8);
            v8f z = {};
            sc[t] = wmma_bf16(qf, kf, z);
        }
        const float* brow = biasH + ((size_t)h * 256 + qbase) * 256 + m16;
        float mrow[8];
#pragma unroll
        for (int r = 0; r < 8; ++r) mrow[r] = -3.0e38f;
#pragma unroll
        for (int t = 0; t < 16; ++t) {
#pragma unroll
            for (int r = 0; r < 8; ++r) {
                const int q = (lane < 16) ? r : r + 8;
                sc[t][r] = sc[t][r] * scale + brow[(size_t)q * 256 + t * 16];
                mrow[r] = fmaxf(mrow[r], sc[t][r]);
            }
        }
        // ---- one cross-lane max reduction per row ----
#pragma unroll
        for (int r = 0; r < 8; ++r) {
            float mx = mrow[r];
            mx = fmaxf(mx, __shfl_xor(mx, 1));
            mx = fmaxf(mx, __shfl_xor(mx, 2));
            mx = fmaxf(mx, __shfl_xor(mx, 4));
            mx = fmaxf(mx, __shfl_xor(mx, 8));
            mrow[r] = mx;
        }
        // ---- exp + per-lane sum ----
        float lsum[8];
#pragma unroll
        for (int r = 0; r < 8; ++r) lsum[r] = 0.0f;
#pragma unroll
        for (int t = 0; t < 16; ++t) {
#pragma unroll
            for (int r = 0; r < 8; ++r) {
                const float p = __expf(sc[t][r] - mrow[r]);
                sc[t][r] = p;
                lsum[r] += p;
            }
        }
#pragma unroll
        for (int r = 0; r < 8; ++r) {
            float sm = lsum[r];
            sm += __shfl_xor(sm, 1);
            sm += __shfl_xor(sm, 2);
            sm += __shfl_xor(sm, 4);
            sm += __shfl_xor(sm, 8);
            lsum[r] = sm;
        }

        // ---- pack full P block (C-layout -> row-major LDS) ----
        bf16* ps = Pscr[wave];
#pragma unroll
        for (int t = 0; t < 16; ++t) {
#pragma unroll
            for (int r = 0; r < 8; ++r) {
                const int q = (lane < 16) ? r : r + 8;
                ps[q * 256 + t * 16 + m16] = (bf16)sc[t][r];
            }
        }
        asm volatile("s_wait_dscnt 0" ::: "memory");

        // ---- phase 2: all 16 AV WMMAs ----
        v8f acc0 = {}, acc1 = {};
#pragma unroll
        for (int kc = 0; kc < 8; ++kc) {
            Frag pa;
            const bf16* pp = ps + m16 * 256 + kc * 32 + ((lane < 16) ? 0 : 16);
            pa.u[0] = *(const v4u*)(pp);
            pa.u[1] = *(const v4u*)(pp + 8);
            Frag vf0, vf1;
            const int kofs = kc * 32 + ((lane < 16) ? 0 : 16);
            const bf16* v0 = VT + m16 * 256 + kofs;
            vf0.u[0] = *(const v4u*)(v0);
            vf0.u[1] = *(const v4u*)(v0 + 8);
            const bf16* v1 = VT + (16 + m16) * 256 + kofs;
            vf1.u[0] = *(const v4u*)(v1);
            vf1.u[1] = *(const v4u*)(v1 + 8);
            acc0 = wmma_bf16(pa, vf0, acc0);
            acc1 = wmma_bf16(pa, vf1, acc1);
        }

        // ---- epilogue: normalize and store [b*256+q, h*32+d] as bf16 ----
        bf16* orow = attn_out + ((size_t)(b * 256 + qbase)) * 512 + h * 32;
#pragma unroll
        for (int r = 0; r < 8; ++r) {
            const int q = (lane < 16) ? r : r + 8;
            const float inv = 1.0f / lsum[r];
            orow[(size_t)q * 512 + m16]      = (bf16)(acc0[r] * inv);
            orow[(size_t)q * 512 + 16 + m16] = (bf16)(acc1[r] * inv);
        }
    }
}

// ---------------------------------------------------------------------------
extern "C" void kernel_launch(void* const* d_in, const int* in_sizes, int n_in,
                              void* d_out, int out_size, void* d_ws, size_t ws_size,
                              hipStream_t stream) {
    const float* x          = (const float*)d_in[0];
    const float* qkv_w      = (const float*)d_in[1];
    const float* qkv_b      = (const float*)d_in[2];
    const float* bias_table = (const float*)d_in[3];
    const float* proj_w     = (const float*)d_in[4];
    const float* proj_b     = (const float*)d_in[5];
    const int*   rel_index  = (const int*)d_in[6];

    const int M = 128 * 256;   // 32768 rows
    char* ws = (char*)d_ws;
    bf16*  xb    = (bf16*)ws;  ws += (size_t)M * 512 * 2;
    bf16*  qwb   = (bf16*)ws;  ws += (size_t)1536 * 512 * 2;
    bf16*  pwb   = (bf16*)ws;  ws += (size_t)512 * 512 * 2;
    float* biasH = (float*)ws; ws += (size_t)16 * 256 * 256 * 4;
    bf16*  qkv   = (bf16*)ws;  ws += (size_t)M * 1536 * 2;
    bf16*  aout  = (bf16*)ws;

    cvt_f32_bf16<<<2048, 256, 0, stream>>>(x, xb, M * 512 / 4);
    cvt_f32_bf16<<<256, 256, 0, stream>>>(qkv_w, qwb, 1536 * 512 / 4);
    cvt_f32_bf16<<<128, 256, 0, stream>>>(proj_w, pwb, 512 * 512 / 4);
    build_bias<<<256, 256, 0, stream>>>(bias_table, rel_index, biasH);

    gemm_wmma<true><<<(M / 64) * (1536 / 256), 256, 0, stream>>>(
        xb, qwb, qkv_b, (void*)qkv, M, 1536, 512);

    attn_wmma<<<128 * 16, 256, 0, stream>>>(qkv, biasH, aout);

    gemm_wmma<false><<<(M / 64) * (512 / 256), 256, 0, stream>>>(
        aout, pwb, proj_b, d_out, M, 512, 512);
}
